// MultiHeadDotProductAttention_10831907520536
// MI455X (gfx1250) — matrix operand order repaired and checked by reading, not compiled, not run on hardware
//
#include <hip/hip_runtime.h>

// MI455X (gfx1250):
//  * all 5 matmul stages on v_wmma_f32_16x16x32_bf16 (fp32 accum)
//  * flash attention (online softmax in exp2 domain, log2e folded into Q scale)
//  * VALU cross-lane reductions via v_permlane16_b32 (no ds_bpermute)
//  * async global->LDS staging (GLOBAL_LOAD_ASYNC_TO_LDS_B128 + s_wait_asynccnt)
//  * weights pre-transposed to bf16 once so all LDS staging is contiguous B128

typedef __attribute__((ext_vector_type(16))) __bf16 v16bf;
typedef __attribute__((ext_vector_type(8)))  float  v8f;
typedef __attribute__((ext_vector_type(4)))  unsigned int v4u;
typedef int v4i_g __attribute__((vector_size(16)));  // matches builtin pointee

union FragU { v16bf v; v4u u[2]; };

#define NEG_BIG (-3.0e38f)
#define LOG2E 1.4426950408889634f

// ------------------------------------------------ feature probes
#if __has_builtin(__builtin_amdgcn_global_load_async_to_lds_b128)
#define HAVE_ASYNC 1
#else
#define HAVE_ASYNC 0
#warning "probe: __builtin_amdgcn_global_load_async_to_lds_b128 NOT available; direct LDS staging"
#endif

#if __has_builtin(__builtin_amdgcn_permlane16)
#define HAVE_PERMLANE 1
#else
#define HAVE_PERMLANE 0
#warning "probe: __builtin_amdgcn_permlane16 NOT available; falling back to __shfl_xor"
#endif

#if __has_builtin(__builtin_amdgcn_exp2f)
#define EXP2F(x) __builtin_amdgcn_exp2f(x)
#else
#define EXP2F(x) exp2f(x)
#warning "probe: __builtin_amdgcn_exp2f NOT available; using exp2f"
#endif

#if HAVE_ASYNC
__device__ __forceinline__ void async_b128(void* lds, const void* g) {
  typedef __attribute__((address_space(1))) v4i_g* gp_t;
  typedef __attribute__((address_space(3))) v4i_g* lp_t;
  __builtin_amdgcn_global_load_async_to_lds_b128(
      (gp_t)(size_t)g, (lp_t)(unsigned int)(size_t)lds, 0, 0);
}
__device__ __forceinline__ void async_wait0() {
#if __has_builtin(__builtin_amdgcn_s_wait_asynccnt)
  __builtin_amdgcn_s_wait_asynccnt(0);
#else
  asm volatile("s_wait_asynccnt 0" ::: "memory");
#endif
}
#endif

// copy 16B global -> LDS (async DMA when available)
__device__ __forceinline__ void stage_b128(unsigned short* lds,
                                           const unsigned short* g) {
#if HAVE_ASYNC
  async_b128(lds, g);
#else
  *(v4u*)lds = *(const v4u*)g;
#endif
}
__device__ __forceinline__ void stage_fence() {
#if HAVE_ASYNC
  async_wait0();
#endif
}

// ------------------------------------------------ lane xor-butterfly (16-lane)
#if HAVE_PERMLANE
__device__ __forceinline__ float lane_perm(float x, unsigned s0, unsigned s1) {
  union { float f; unsigned u; } a, r;
  a.f = x;
  r.u = __builtin_amdgcn_permlane16(a.u, a.u, s0, s1, false, false);
  return r.f;
}
#define XOR1(x) lane_perm((x), 0x67452301u, 0xEFCDAB89u)
#define XOR2(x) lane_perm((x), 0x45670123u, 0xCDEF89ABu)
#define XOR4(x) lane_perm((x), 0x32107654u, 0xBA98FEDCu)
#define XOR8(x) lane_perm((x), 0xFEDCBA98u, 0x76543210u)
#else
#define XOR1(x) __shfl_xor((x), 1)
#define XOR2(x) __shfl_xor((x), 2)
#define XOR4(x) __shfl_xor((x), 4)
#define XOR8(x) __shfl_xor((x), 8)
#endif

__device__ __forceinline__ unsigned short f2bf(float f) {
  union { float f; unsigned int u; } c; c.f = f;
  return (unsigned short)((c.u + 0x7FFFu + ((c.u >> 16) & 1u)) >> 16);  // RNE
}

__device__ __forceinline__ v8f wmma_bf16(v16bf a, v16bf b, v8f c) {
  return __builtin_amdgcn_wmma_f32_16x16x32_bf16(false, a, false, b, (short)0, c,
                                                 false, false);
}

// A fragment (16x32 bf16) from a row-major tile (rs = row stride in u16).
__device__ __forceinline__ v16bf load_afrag(const unsigned short* t, int row0,
                                            int rs, int kofs) {
  int lane = threadIdx.x & 31;
  int lh = lane >> 4, r = lane & 15;
  const unsigned short* p = t + (size_t)(row0 + r) * rs + kofs + lh * 8;
  FragU f;
  f.u[0] = *(const v4u*)(p);
  f.u[1] = *(const v4u*)(p + 16);
  return f.v;
}

// B fragment (32x16 bf16) from a TRANSPOSED tile Bt[n][k] (rs in u16).
__device__ __forceinline__ v16bf load_bfrag(const unsigned short* t, int n0,
                                            int rs, int kofs) {
  int lane = threadIdx.x & 31;
  int lh = lane >> 4, n = lane & 15;
  const unsigned short* p = t + (size_t)(n0 + n) * rs + kofs + lh * 8;
  FragU f;
  f.u[0] = *(const v4u*)(p);
  f.u[1] = *(const v4u*)(p + 16);
  return f.v;
}

// ---------------------------------------------------------------- converts
__global__ void cvt_bf16_kernel(const float* __restrict__ in,
                                unsigned short* __restrict__ out, int n) {
  int i = blockIdx.x * blockDim.x + threadIdx.x;
  if (i < n) out[i] = f2bf(in[i]);
}

// fp32 [1024][1024] -> bf16 transposed [n][k], LDS-tiled for coalescing
__global__ __launch_bounds__(256) void cvt_wt_kernel(
    const float* __restrict__ in, unsigned short* __restrict__ out) {
  __shared__ float tile[32][33];
  const int k0 = blockIdx.x * 32, n0 = blockIdx.y * 32;
  const int tx = threadIdx.x & 31, ty = threadIdx.x >> 5;
#pragma unroll
  for (int i = 0; i < 32; i += 8)
    tile[ty + i][tx] = in[(size_t)(k0 + ty + i) * 1024 + n0 + tx];
  __syncthreads();
#pragma unroll
  for (int i = 0; i < 32; i += 8)
    out[(size_t)(n0 + ty + i) * 1024 + k0 + tx] = f2bf(tile[tx][ty + i]);
}

// ---------------------------------------------------------------- GEMM
// Y[M x 1024] = X[M x 1024]_bf16 * W (given as Wt[n][k] bf16) ; epilogue:
//   MODE 0: y=(acc+bias)*scale -> bf16 scattered to [B,H,S,64] (QKV proj)
//   MODE 1: y=(acc+bias)       -> fp32 [M x 1024]              (out proj)
template <int MODE>
__global__ __launch_bounds__(256) void gemm_bf16_kernel(
    const unsigned short* __restrict__ X, const unsigned short* __restrict__ Wt_g,
    const float* __restrict__ bias, void* __restrict__ out, int Mtotal,
    float scale) {
  __shared__ __align__(16) unsigned short Xt[128 * 32];  // [row][k]
  __shared__ __align__(16) unsigned short Wt[128 * 32];  // [n][k]
  const int t = threadIdx.x;
  const int wave = t >> 5;
  const int lane = t & 31, lh = lane >> 4, cn = lane & 15;
  const int m0 = blockIdx.x * 128;
  const int n0 = blockIdx.y * 128;

  v8f acc[8];
#pragma unroll
  for (int i = 0; i < 8; ++i) acc[i] = (v8f)0.0f;

  for (int kk = 0; kk < 1024; kk += 32) {
    __syncthreads();
    {  // stage X tile: 128 rows x 32 k
      int row = t >> 1;
      int ks = (t & 1) * 16;
      int grow = min(m0 + row, Mtotal - 1);
      const unsigned short* src = X + (size_t)grow * 1024 + kk + ks;
      unsigned short* dst = &Xt[row * 32 + ks];
      stage_b128(dst, src);
      stage_b128(dst + 8, src + 8);
    }
    {  // stage Wt tile: 128 n-rows x 32 k (weights pre-transposed -> contiguous)
#pragma unroll
      for (int l = 0; l < 2; ++l) {
        int idx = t * 2 + l;       // 0..511
        int row = idx >> 2;        // 0..127
        int seg = (idx & 3) * 8;   // 0,8,16,24
        stage_b128(&Wt[row * 32 + seg],
                   Wt_g + (size_t)(n0 + row) * 1024 + kk + seg);
      }
    }
    stage_fence();
    __syncthreads();
    v16bf af = load_afrag(Xt, wave * 16, 32, 0);
#pragma unroll
    for (int s = 0; s < 8; ++s) {
      v16bf bfr = load_bfrag(Wt, s * 16, 32, 0);
      acc[s] = wmma_bf16(af, bfr, acc[s]);
    }
  }

  // epilogue
#pragma unroll
  for (int s = 0; s < 8; ++s) {
    int col = n0 + s * 16 + cn;
    float bv = bias[col];
#pragma unroll
    for (int r = 0; r < 8; ++r) {
      int row = m0 + wave * 16 + lh * 8 + r;
      if (row < Mtotal) {
        float y = (acc[s][r] + bv) * scale;
        if (MODE == 0) {
          int bb = row / 1500, sidx = row % 1500;
          int h = col >> 6, hd = col & 63;
          ((unsigned short*)out)[(((size_t)(bb * 16 + h) * 1500) + sidx) * 64 +
                                 hd] = f2bf(y);
        } else {
          ((float*)out)[(size_t)row * 1024 + col] = y;
        }
      }
    }
  }
}

// ---------------------------------------------------------------- attention
// Per (b,h): flash attention over S=1500 keys, HD=64. Workgroup = 128 query
// rows (8 waves x 16). K tile = 32 keys (K row-major via async DMA, V^T via
// VGPR transpose). Scores are in the exp2 domain (log2e folded into Q).
__global__ __launch_bounds__(256) void flash_attn_kernel(
    const unsigned short* __restrict__ Qw,   // [B*H][1500][64] bf16 (pre-scaled)
    const unsigned short* __restrict__ Kw,   // [B*H][1500][64] bf16
    const unsigned short* __restrict__ Vw,   // [B*H][1500][64] bf16
    unsigned short* __restrict__ Ow) {       // [B*1500][1024] bf16
  __shared__ __align__(16) unsigned short Kt[32 * 64];     // [key][hd]
  __shared__ __align__(16) unsigned short Vt[64 * 32];     // [hd][key]
  __shared__ __align__(16) unsigned short Pb[8][16 * 32];  // per-wave P tile

  const int t = threadIdx.x;
  const int wave = t >> 5;
  const int lane = t & 31, lh = lane >> 4, cn = lane & 15;
  const int bh = blockIdx.y;  // 0..63
  const int h = bh & 15, bb = bh >> 4;
  const int q0 = blockIdx.x * 128 + wave * 16;
  const size_t base = (size_t)bh * 1500 * 64;

  // Q fragments (this wave's 16 rows, k-chunks 0..31 / 32..63), kept in regs.
  v16bf qa0, qa1;
  {
    int row = min(q0 + (lane & 15), 1499);
    const unsigned short* p = Qw + base + (size_t)row * 64;
    FragU f0, f1;
    f0.u[0] = *(const v4u*)(p + lh * 8);
    f0.u[1] = *(const v4u*)(p + 16 + lh * 8);
    f1.u[0] = *(const v4u*)(p + 32 + lh * 8);
    f1.u[1] = *(const v4u*)(p + 48 + lh * 8);
    qa0 = f0.v;
    qa1 = f1.v;
  }

  v8f accO[4];
#pragma unroll
  for (int i = 0; i < 4; ++i) accO[i] = (v8f)0.0f;
  float m8[8], l8[8];
#pragma unroll
  for (int r = 0; r < 8; ++r) {
    m8[r] = NEG_BIG;
    l8[r] = 0.0f;
  }

  const int nkt = (1500 + 31) / 32;  // 47
  for (int kt = 0; kt < nkt; ++kt) {
    const int kbase = kt * 32;
    __syncthreads();
    {  // stage K (async DMA, row-major) and V (VGPR transpose): 32 keys x 64 hd
      int row = t >> 3;      // 0..31
      int c0 = (t & 7) * 8;  // 0..56
      size_t off = base + (size_t)(kbase + row) * 64 + c0;
      // overreads past 1500 stay inside the workspace; results are masked
      stage_b128(&Kt[row * 64 + c0], Kw + off);
      v4u dv = *(const v4u*)(Vw + off);
      const unsigned short* e = (const unsigned short*)&dv;
#pragma unroll
      for (int j = 0; j < 8; ++j) Vt[(c0 + j) * 32 + row] = e[j];
      if (kt + 1 < nkt)  // prefetch next V tile (global_prefetch_b8)
        __builtin_prefetch(Vw + off + 32 * 64, 0, 1);
    }
    stage_fence();
    __syncthreads();

    // S = Q * K^T : two 16x16 subtiles over this wave's rows
    v8f accS[2];
#pragma unroll
    for (int s = 0; s < 2; ++s) {
      v8f a = (v8f)0.0f;
      a = wmma_bf16(qa0, load_bfrag(Kt, s * 16, 64, 0), a);
      a = wmma_bf16(qa1, load_bfrag(Kt, s * 16, 64, 32), a);
      int col = kbase + s * 16 + cn;
      if (col >= 1500) {
#pragma unroll
        for (int r = 0; r < 8; ++r) a[r] = NEG_BIG;
      }
      accS[s] = a;
    }

    // online softmax: row stats in element r (row lh*8+r); the row's 16 cols
    // live in one 16-lane half -> v_permlane16 xor butterfly
#pragma unroll
    for (int r = 0; r < 8; ++r) {
      float x = fmaxf(accS[0][r], accS[1][r]);
      x = fmaxf(x, XOR1(x));
      x = fmaxf(x, XOR2(x));
      x = fmaxf(x, XOR4(x));
      x = fmaxf(x, XOR8(x));
      float mn = fmaxf(m8[r], x);
      float alpha = EXP2F(m8[r] - mn);
      m8[r] = mn;
      float p0 = EXP2F(accS[0][r] - mn);
      float p1 = EXP2F(accS[1][r] - mn);
      accS[0][r] = p0;
      accS[1][r] = p1;
      float sum = p0 + p1;
      sum += XOR1(sum);
      sum += XOR2(sum);
      sum += XOR4(sum);
      sum += XOR8(sum);
      l8[r] = l8[r] * alpha + sum;
#pragma unroll
      for (int n = 0; n < 4; ++n) accO[n][r] = accO[n][r] * alpha;
    }

    // P (C-layout) -> per-wave LDS row-major [16][32] -> A fragment
    unsigned short* pb = Pb[wave];
#pragma unroll
    for (int s = 0; s < 2; ++s)
#pragma unroll
      for (int r = 0; r < 8; ++r)
        pb[(lh * 8 + r) * 32 + s * 16 + cn] = f2bf(accS[s][r]);

    v16bf pf = load_afrag(pb, 0, 32, 0);
#pragma unroll
    for (int n = 0; n < 4; ++n) {
      v16bf vf = load_bfrag(Vt, n * 16, 32, 0);
      accO[n] = wmma_bf16(pf, vf, accO[n]);
    }
  }

  // normalize and emit attention output as bf16 [token][h*64+hd]
#pragma unroll
  for (int r = 0; r < 8; ++r) {
    int q = q0 + lh * 8 + r;
    if (q < 1500) {
      float inv = 1.0f / l8[r];
      size_t rowbase = ((size_t)bb * 1500 + q) * 1024 + h * 64;
#pragma unroll
      for (int n = 0; n < 4; ++n)
        Ow[rowbase + n * 16 + cn] = f2bf(accO[n][r] * inv);
    }
  }
}

// ---------------------------------------------------------------- launch
extern "C" void kernel_launch(void* const* d_in, const int* in_sizes, int n_in,
                              void* d_out, int out_size, void* d_ws,
                              size_t ws_size, hipStream_t stream) {
  const float* inputs_q = (const float*)d_in[0];
  const float* inputs_kv = (const float*)d_in[1];
  const float* Wq = (const float*)d_in[2];
  const float* bq = (const float*)d_in[3];
  const float* Wk = (const float*)d_in[4];
  const float* bk = (const float*)d_in[5];
  const float* Wv = (const float*)d_in[6];
  const float* bv = (const float*)d_in[7];
  const float* Wo = (const float*)d_in[8];
  const float* bo = (const float*)d_in[9];

  const size_t TOK = 6000ull * 1024ull;  // B*S tokens x D
  const size_t WEL = 1024ull * 1024ull;  // weight elements
  unsigned short* xq = (unsigned short*)d_ws;
  unsigned short* xkv = xq + TOK;
  unsigned short* wqb = xkv + TOK;  // transposed [n][k]
  unsigned short* wkb = wqb + WEL;
  unsigned short* wvb = wkb + WEL;
  unsigned short* wob = wvb + WEL;
  unsigned short* qw = wob + WEL;  // [B,H,1500,64]
  unsigned short* kw = qw + TOK;
  unsigned short* vw = kw + TOK;
  unsigned short* ao = vw + TOK;  // [B*1500,1024]

  const int nt = (int)TOK;
  cvt_bf16_kernel<<<(nt + 255) / 256, 256, 0, stream>>>(inputs_q, xq, nt);
  cvt_bf16_kernel<<<(nt + 255) / 256, 256, 0, stream>>>(inputs_kv, xkv, nt);
  dim3 gt(32, 32);
  cvt_wt_kernel<<<gt, 256, 0, stream>>>(Wq, wqb);
  cvt_wt_kernel<<<gt, 256, 0, stream>>>(Wk, wkb);
  cvt_wt_kernel<<<gt, 256, 0, stream>>>(Wv, wvb);
  cvt_wt_kernel<<<gt, 256, 0, stream>>>(Wo, wob);

  dim3 gg((6000 + 127) / 128, 8);
  // q gets the folded 1/sqrt(64) scale, plus log2e for exp2-domain softmax
  gemm_bf16_kernel<0><<<gg, 256, 0, stream>>>(xq, wqb, bq, qw, 6000,
                                              0.125f * LOG2E);
  gemm_bf16_kernel<0><<<gg, 256, 0, stream>>>(xkv, wkb, bk, kw, 6000, 1.0f);
  gemm_bf16_kernel<0><<<gg, 256, 0, stream>>>(xkv, wvb, bv, vw, 6000, 1.0f);

  dim3 ga((1500 + 127) / 128, 64);
  flash_attn_kernel<<<ga, 256, 0, stream>>>(qw, kw, vw, ao);

  gemm_bf16_kernel<1><<<gg, 256, 0, stream>>>(ao, wob, bo, d_out, 6000, 1.0f);
}
